// MoaiSSM_29463475650729
// MI455X (gfx1250) — matrix operand order repaired and checked by reading, not compiled, not run on hardware
//
#include <hip/hip_runtime.h>
#include <hip/hip_bf16.h>

typedef __attribute__((ext_vector_type(16))) _Float16 v16h;
typedef __attribute__((ext_vector_type(8)))  float    v8f;

#define D_MODEL 768
#define D_STATE 16
#define D_CONV  4
#define D_INNER 1536
#define BATCH   2
#define SEQLEN  4096
#define NROWS   (BATCH * SEQLEN)   // 8192

// ---------------------------------------------------------------------------
// Pack fp32 weight W[K,N] into f16 WMMA B-fragments, N padded to Npad (zeros).
// Fragment order: frag = nt*(K/32)+kb ; per lane 16 contiguous f16 (32B).
// B layout per ISA: lane l16 = column nt*16+l16; elems 0..7 -> K = kb*32+hi*8+e ; elems 8..15 -> +16.
// ---------------------------------------------------------------------------
__global__ __launch_bounds__(256) void ssm_pack_b(
    const float* __restrict__ W, _Float16* __restrict__ P,
    int K, int N, int Npad)
{
    const int kf = K >> 5;
    const int t  = blockIdx.x * 256 + threadIdx.x;
    const int total = (Npad >> 4) * kf * 32;
    if (t >= total) return;
    const int lane = t & 31;
    const int frag = t >> 5;
    const int kb   = frag % kf;
    const int nt   = frag / kf;
    const int l16  = lane & 15;
    const int hi   = lane >> 4;
    const int n    = nt * 16 + l16;
    v16h v;
#pragma unroll
    for (int e = 0; e < 16; ++e) {
        const int k = kb * 32 + hi * 8 + (e & 7) + ((e >> 3) << 4);
        v[e] = (n < N) ? (_Float16)W[(size_t)k * N + n] : (_Float16)0.f;
    }
    *(v16h*)(P + (size_t)t * 16) = v;
}

// ---------------------------------------------------------------------------
// Pack fp32 activation A[M,K] (leading dim lda) into f16 WMMA A-fragments.
// Fragment order: frag = mt*(K/32)+kb ; per lane 16 contiguous f16 (32B).
// A layout per ISA: lane l16 = row mt*16+l16;
//   elems 0..7 -> K = kb*32 + hi*8 + e ; elems 8..15 -> +16.
// ---------------------------------------------------------------------------
__global__ __launch_bounds__(256) void ssm_pack_a(
    const float* __restrict__ A, _Float16* __restrict__ P,
    int M, int K, int lda)
{
    const int kf = K >> 5;
    const int t  = blockIdx.x * 256 + threadIdx.x;
    const int total = (M >> 4) * kf * 32;
    if (t >= total) return;
    const int lane = t & 31;
    const int frag = t >> 5;
    const int kb   = frag % kf;
    const int mt   = frag / kf;
    const int l16  = lane & 15;
    const int hi   = lane >> 4;
    const float* ar = A + (size_t)(mt * 16 + l16) * lda + kb * 32 + hi * 8;
    v16h v;
#pragma unroll
    for (int e = 0; e < 8; ++e) {
        v[e]     = (_Float16)ar[e];
        v[8 + e] = (_Float16)ar[16 + e];
    }
    *(v16h*)(P + (size_t)t * 16) = v;
}

// ---------------------------------------------------------------------------
// WMMA GEMM on pre-packed f16 fragments: C[M,N] = A @ B, fp32 accumulation.
// Each wave computes a 32(M) x 64(N) strip: 8 v_wmma per K-step against
// 6 contiguous 32-byte fragment loads. Branch-free inner loop (Npad-padded B).
// EPI 0: plain store.  EPI 1: softplus(acc + bias[col]).
// ---------------------------------------------------------------------------
template <int EPI>
__global__ __launch_bounds__(256) void ssm_gemm_wmma(
    const _Float16* __restrict__ Ap, const _Float16* __restrict__ Bp,
    const float* __restrict__ bias, float* __restrict__ C,
    int M, int N, int Npad, int K, int ldc)
{
    const int wave = threadIdx.x >> 5;
    const int lane = threadIdx.x & 31;
    const int l16  = lane & 15;
    const int hi   = lane >> 4;

    const int kf    = K >> 5;
    const int ntn64 = Npad >> 6;
    const int st = blockIdx.x * 8 + wave;
    const int mt = st / ntn64;           // 32-row super tile
    const int ns = st % ntn64;
    const int m0 = mt * 32;
    if (m0 >= M) return;                 // wave-uniform exit
    const int n0 = ns * 64;

    const _Float16* a0p = Ap + (size_t)(mt * 2)     * kf * 512 + lane * 16;
    const _Float16* a1p = Ap + (size_t)(mt * 2 + 1) * kf * 512 + lane * 16;
    const _Float16* bpb = Bp + (size_t)(n0 >> 4)    * kf * 512 + lane * 16;
    const size_t bstride = (size_t)kf * 512;       // per 16-col tile

    v8f acc[2][4] = {};
    for (int kb = 0; kb < kf; ++kb) {
        __builtin_prefetch(a0p + 2048, 0, 1);
        const v16h a0 = *(const v16h*)a0p;  a0p += 512;
        const v16h a1 = *(const v16h*)a1p;  a1p += 512;
        const _Float16* bk = bpb + (size_t)kb * 512;
#pragma unroll
        for (int j = 0; j < 4; ++j) {
            const v16h bf = *(const v16h*)(bk + (size_t)j * bstride);
            acc[0][j] = __builtin_amdgcn_wmma_f32_16x16x32_f16(
                false, a0, false, bf, (short)0, acc[0][j], false, false);
            acc[1][j] = __builtin_amdgcn_wmma_f32_16x16x32_f16(
                false, a1, false, bf, (short)0, acc[1][j], false, false);
        }
    }

#pragma unroll
    for (int i = 0; i < 2; ++i) {
#pragma unroll
        for (int j = 0; j < 4; ++j) {
            const int n = n0 + j * 16;
            if (n < N) {                               // uniform (N % 16 == 0)
                const int col = n + l16;
#pragma unroll
                for (int r = 0; r < 8; ++r) {
                    const int row = m0 + i * 16 + r + 8 * hi;  // C/D layout
                    float v = acc[i][j][r];
                    if (EPI == 1) {
                        v += bias[col];
                        v = (v > 20.f) ? v : __logf(1.f + __expf(v));
                    }
                    C[(size_t)row * ldc + col] = v;
                }
            }
        }
    }
}

// ---------------------------------------------------------------------------
// Depthwise causal conv (width 4) + bias + SiLU.  Reads x = xz[:, :D_INNER].
// ---------------------------------------------------------------------------
__global__ __launch_bounds__(256) void ssm_conv_silu(
    const float* __restrict__ xz, const float* __restrict__ cw,
    const float* __restrict__ cb, float* __restrict__ xc)
{
    const long i = (long)blockIdx.x * blockDim.x + threadIdx.x;
    const long total = (long)NROWS * D_INNER;
    if (i >= total) return;
    const int  d = (int)(i % D_INNER);
    const long r = i / D_INNER;          // r = b*SEQLEN + l
    const int  l = (int)(r % SEQLEN);
    float acc = cb[d];
#pragma unroll
    for (int j = 0; j < D_CONV; ++j) {
        const int ls = l - (D_CONV - 1) + j;
        if (ls >= 0)
            acc += cw[d * D_CONV + j] *
                   xz[(r - (D_CONV - 1) + j) * (long)(2 * D_INNER) + d];
    }
    xc[r * (long)D_INNER + d] = acc / (1.f + __expf(-acc));   // SiLU
}

// ---------------------------------------------------------------------------
// Per-row softmax over first 16 cols of xp -> Bm ; copy cols 16..31 -> Cm.
// ---------------------------------------------------------------------------
__global__ __launch_bounds__(256) void ssm_softmax_bc(
    const float* __restrict__ xp, float* __restrict__ Bm, float* __restrict__ Cm)
{
    const int row  = blockIdx.x * 8 + (threadIdx.x >> 5);
    const int lane = threadIdx.x & 31;
    if (row >= NROWS) return;
    const float* p = xp + (long)row * (2 * D_STATE + D_INNER);
    if (lane < 16) {
        float v = p[lane];
        float m = v;
        for (int o = 8; o; o >>= 1) m = fmaxf(m, __shfl_xor(m, o, 16));
        float e = __expf(v - m);
        float s = e;
        for (int o = 8; o; o >>= 1) s += __shfl_xor(s, o, 16);
        Bm[row * D_STATE + lane] = e / s;
        Cm[row * D_STATE + lane] = p[D_STATE + lane];
    }
}

// ---------------------------------------------------------------------------
// Selective scan: one channel (b,d) per thread, 16-state recurrence in
// registers, B_t / C_t broadcast through LDS each timestep.
// ---------------------------------------------------------------------------
__global__ __launch_bounds__(256) void ssm_scan(
    const float* __restrict__ delta, const float* __restrict__ Bm,
    const float* __restrict__ Cm,    const float* __restrict__ xc,
    const float* __restrict__ A_log, float* __restrict__ y)
{
    __shared__ float sB[D_STATE];
    __shared__ float sC[D_STATE];
    const int b = blockIdx.x / (D_INNER / 256);
    const int d = (blockIdx.x % (D_INNER / 256)) * 256 + threadIdx.x;

    float Ac[D_STATE], h[D_STATE];
#pragma unroll
    for (int s = 0; s < D_STATE; ++s) {
        Ac[s] = -__expf(A_log[d * D_STATE + s]);
        h[s]  = 0.f;
    }
    for (int t = 0; t < SEQLEN; ++t) {
        const long row = (long)b * SEQLEN + t;
        if (threadIdx.x < 16)       sB[threadIdx.x]      = Bm[row * D_STATE + threadIdx.x];
        else if (threadIdx.x < 32)  sC[threadIdx.x - 16] = Cm[row * D_STATE + threadIdx.x - 16];
        __syncthreads();
        const float dt = delta[row * D_INNER + d];
        const float du = dt * xc[row * D_INNER + d];
        float yv = 0.f;
#pragma unroll
        for (int s = 0; s < D_STATE; ++s) {
            const float dA = __expf(dt * Ac[s]);
            h[s] = dA * h[s] + du * sB[s];
            yv  += sC[s] * h[s];
        }
        y[row * D_INNER + d] = yv;
        __syncthreads();
    }
}

// ---------------------------------------------------------------------------
// In-place gating: y *= silu(z),  z = xz[:, D_INNER:]
// ---------------------------------------------------------------------------
__global__ __launch_bounds__(256) void ssm_gate(
    float* __restrict__ y, const float* __restrict__ xz)
{
    const long i = (long)blockIdx.x * blockDim.x + threadIdx.x;
    if (i >= (long)NROWS * D_INNER) return;
    const long row = i / D_INNER;
    const int  d   = (int)(i % D_INNER);
    const float z  = xz[row * (long)(2 * D_INNER) + D_INNER + d];
    y[i] *= z / (1.f + __expf(-z));
}

// ---------------------------------------------------------------------------
extern "C" void kernel_launch(void* const* d_in, const int* in_sizes, int n_in,
                              void* d_out, int out_size, void* d_ws, size_t ws_size,
                              hipStream_t stream)
{
    const float* hidden = (const float*)d_in[0];  // [B,L,768]
    const float* W_in   = (const float*)d_in[1];  // [768, 3072]
    const float* conv_w = (const float*)d_in[2];  // [1536,1,4]
    const float* conv_b = (const float*)d_in[3];  // [1536]
    const float* W_x    = (const float*)d_in[4];  // [1536, 1568]
    const float* W_dt   = (const float*)d_in[5];  // [1536, 1536]
    const float* b_dt   = (const float*)d_in[6];  // [1536]
    const float* A_log  = (const float*)d_in[7];  // [1536, 16]
    const float* W_out  = (const float*)d_in[9];  // [1536, 768]
    float* out = (float*)d_out;                   // [B,L,768]

    // fp32 workspace regions
    float* ws = (float*)d_ws;
    float* xz   = ws;                                        // [8192, 3072]
    float* xc   = xz  + (long)NROWS * 2 * D_INNER;           // [8192, 1536]
    float* xp   = xc  + (long)NROWS * D_INNER;               // [8192, 1568]
    float* dl   = xp  + (long)NROWS * (2 * D_STATE + D_INNER); // [8192, 1536]
    float* Bm   = dl  + (long)NROWS * D_INNER;               // [8192, 16]
    float* Cm   = Bm  + (long)NROWS * D_STATE;               // [8192, 16]
    float* ybuf = Cm  + (long)NROWS * D_STATE;               // [8192, 1536]
    float* fend = ybuf + (long)NROWS * D_INNER;

    // f16 packed-fragment staging (reused across the sequential GEMMs)
    _Float16* PA = (_Float16*)fend;                          // max 8192x1536
    _Float16* PB = PA + (size_t)NROWS * D_INNER;             // max 48*100*512

    auto packb_grid = [](int K, int Npad) {
        return ((Npad >> 4) * (K >> 5) * 32 + 255) / 256;
    };
    auto packa_grid = [](int M, int K) {
        return ((M >> 4) * (K >> 5) * 32 + 255) / 256;
    };
    auto gemm_grid = [](int M, int Npad) {
        long tiles = (long)(M / 32) * (Npad / 64);
        return (int)((tiles + 7) / 8);
    };
    const long ew_total = (long)NROWS * D_INNER;
    const int  ew_grid  = (int)((ew_total + 255) / 256);

    // 1) xz = hidden @ W_in   (8192 x 3072, K=768)
    ssm_pack_b<<<packb_grid(D_MODEL, 3072), 256, 0, stream>>>(W_in, PB, D_MODEL, 3072, 3072);
    ssm_pack_a<<<packa_grid(NROWS, D_MODEL), 256, 0, stream>>>(hidden, PA, NROWS, D_MODEL, D_MODEL);
    ssm_gemm_wmma<0><<<gemm_grid(NROWS, 3072), 256, 0, stream>>>(
        PA, PB, nullptr, xz, NROWS, 3072, 3072, D_MODEL, 3072);

    // 2) depthwise causal conv + bias + SiLU
    ssm_conv_silu<<<ew_grid, 256, 0, stream>>>(xz, conv_w, conv_b, xc);

    // 3) xp = xc @ W_x   (8192 x 1568, K=1536, Npad=1600)
    ssm_pack_b<<<packb_grid(D_INNER, 1600), 256, 0, stream>>>(W_x, PB, D_INNER, 1568, 1600);
    ssm_pack_a<<<packa_grid(NROWS, D_INNER), 256, 0, stream>>>(xc, PA, NROWS, D_INNER, D_INNER);
    ssm_gemm_wmma<0><<<gemm_grid(NROWS, 1600), 256, 0, stream>>>(
        PA, PB, nullptr, xp, NROWS, 1568, 1600, D_INNER, 1568);

    // 4) Bm = softmax(xp[:, :16]),  Cm = xp[:, 16:32]
    ssm_softmax_bc<<<NROWS / 8, 256, 0, stream>>>(xp, Bm, Cm);

    // 5) delta = softplus(xp[:, 32:] @ W_dt + b_dt)   (8192 x 1536, K=1536)
    ssm_pack_b<<<packb_grid(D_INNER, 1536), 256, 0, stream>>>(W_dt, PB, D_INNER, 1536, 1536);
    ssm_pack_a<<<packa_grid(NROWS, D_INNER), 256, 0, stream>>>(
        xp + 2 * D_STATE, PA, NROWS, D_INNER, 2 * D_STATE + D_INNER);
    ssm_gemm_wmma<1><<<gemm_grid(NROWS, 1536), 256, 0, stream>>>(
        PA, PB, b_dt, dl, NROWS, 1536, 1536, D_INNER, 1536);

    // 6) selective scan -> ybuf
    ssm_scan<<<BATCH * (D_INNER / 256), 256, 0, stream>>>(dl, Bm, Cm, xc, A_log, ybuf);

    // 7) ybuf *= silu(z)
    ssm_gate<<<ew_grid, 256, 0, stream>>>(ybuf, xz);

    // 8) out = ybuf @ W_out   (8192 x 768, K=1536)
    ssm_pack_b<<<packb_grid(D_INNER, 768), 256, 0, stream>>>(W_out, PB, D_INNER, 768, 768);
    ssm_pack_a<<<packa_grid(NROWS, D_INNER), 256, 0, stream>>>(ybuf, PA, NROWS, D_INNER, D_INNER);
    ssm_gemm_wmma<0><<<gemm_grid(NROWS, 768), 256, 0, stream>>>(
        PA, PB, nullptr, out, NROWS, D_MODEL, 768, D_INNER, D_MODEL);
}
